// Tactician_64570538328620
// MI455X (gfx1250) — 8-bit matrix operand order repaired and checked by reading, compiled, not run on hardware
//
#include <hip/hip_runtime.h>
#include <math.h>


typedef __attribute__((ext_vector_type(8))) int v8i;

namespace {
constexpr int   D_    = 2048;
constexpr int   NTOK_ = 2048;
constexpr int   H_    = 8192;
constexpr float EPS_  = 1e-5f;

constexpr size_t OFF_WSUM = 0;
constexpr size_t OFF_RS   = 1024;
constexpr size_t OFF_AQ   = 16384;
constexpr size_t OFF_U    = OFF_AQ + (size_t)NTOK_ * H_;
constexpr size_t OFF_H    = OFF_U + (size_t)NTOK_ * H_ * 4;
constexpr size_t OFF_WQ   = OFF_H + (size_t)NTOK_ * D_ * 4;
}

__global__ void zero_kernel(float* __restrict__ p, int n) {
    int i = blockIdx.x * blockDim.x + threadIdx.x;
    if (i < n) p[i] = 0.0f;
}

__global__ void absum_kernel(const float* __restrict__ w, long long n,
                             float* __restrict__ acc) {
    __shared__ float sd[256];
    float s = 0.0f;
    for (long long i = (long long)blockIdx.x * blockDim.x + threadIdx.x; i < n;
         i += (long long)gridDim.x * blockDim.x)
        s += fabsf(w[i]);
    sd[threadIdx.x] = s;
    __syncthreads();
    for (int o = 128; o > 0; o >>= 1) {
        if (threadIdx.x < o) sd[threadIdx.x] += sd[threadIdx.x + o];
        __syncthreads();
    }
    if (threadIdx.x == 0) atomicAdd(acc, sd[0]);
}

__global__ void wquant_kernel(const float* __restrict__ w,
                              signed char* __restrict__ wq, long long n,
                              const float* __restrict__ acc, float inv_n) {
    const float m = fmaxf(acc[0] * inv_n, EPS_);
    const float s = 1.0f / m;
    for (long long i = (long long)blockIdx.x * blockDim.x + threadIdx.x; i < n;
         i += (long long)gridDim.x * blockDim.x) {
        float t = rintf(w[i] * s);
        t = fminf(fmaxf(t, -1.0f), 1.0f);
        wq[i] = (signed char)(int)t;
    }
}

template <int NSRC>
__global__ void actquant_kernel(const float* __restrict__ a0,
                                const float* __restrict__ a1,
                                const float* __restrict__ a2, int Kper,
                                signed char* __restrict__ out,
                                float* __restrict__ rs) {
    __shared__ float sd[256];
    __shared__ float sscale;
    const int row = blockIdx.x;
    const float* base[3];
    base[0] = a0 + (long long)row * Kper;
    base[1] = (NSRC > 1) ? (a1 + (long long)row * Kper) : a0;
    base[2] = (NSRC > 2) ? (a2 + (long long)row * Kper) : a0;

    float amax = 0.0f;
#pragma unroll
    for (int src = 0; src < NSRC; ++src)
        for (int j = threadIdx.x; j < Kper; j += blockDim.x)
            amax = fmaxf(amax, fabsf(base[src][j]));
    sd[threadIdx.x] = amax;
    __syncthreads();
    for (int o = 128; o > 0; o >>= 1) {
        if (threadIdx.x < o) sd[threadIdx.x] = fmaxf(sd[threadIdx.x], sd[threadIdx.x + o]);
        __syncthreads();
    }
    if (threadIdx.x == 0) {
        float am = fmaxf(sd[0], EPS_);
        sscale = 127.0f / am;
        rs[row] = am / 127.0f;
    }
    __syncthreads();
    const float sc = sscale;
    signed char* orow = out + (long long)row * (NSRC * Kper);
#pragma unroll
    for (int src = 0; src < NSRC; ++src)
        for (int j = threadIdx.x; j < Kper; j += blockDim.x) {
            float q = rintf(base[src][j] * sc);
            q = fminf(fmaxf(q, -128.0f), 127.0f);
            orow[src * Kper + j] = (signed char)(int)q;
        }
}

__device__ __forceinline__ v8i load_a_frag(const signed char* base, int h) {
    v8i r;
#pragma unroll
    for (int g = 0; g < 4; ++g) {
        uint2 d = *(const uint2*)(base + g * 16 + h * 8);
        r[2 * g + 0] = (int)d.x;
        r[2 * g + 1] = (int)d.y;
    }
    return r;
}
__device__ __forceinline__ v8i load_b_frag(const signed char* base, int h) {
    v8i r;
#pragma unroll
    for (int g = 0; g < 4; ++g) {
        uint2 d = *(const uint2*)(base + g * 16 + h * 8);
        r[2 * g + 0] = (int)d.x;
        r[2 * g + 1] = (int)d.y;
    }
    return r;
}

template <int MODE>
__global__ __launch_bounds__(256)
void gemm_iu8_kernel(const signed char* __restrict__ A,
                     const signed char* __restrict__ W,
                     const float* __restrict__ rs,
                     const float* __restrict__ wsum, float inv_wn,
                     const float* __restrict__ bias,
                     const float* __restrict__ aux0,
                     const float* __restrict__ aux1,
                     float* __restrict__ out, int N, int K) {
    const int tid  = threadIdx.x;
    const int lane = tid & 31;
    const int wave = tid >> 5;
    const int h    = lane >> 4;
    const int l15  = lane & 15;
    const int wm   = wave >> 2;
    const int wn   = wave & 3;
    const int mbase = blockIdx.y * 32 + wm * 16;
    const int nbase = blockIdx.x * 256 + wn * 64;

    const float mW = fmaxf(wsum[0] * inv_wn, EPS_);

    v8i acc[4];
#pragma unroll
    for (int t = 0; t < 4; ++t)
        acc[t] = (v8i){0, 0, 0, 0, 0, 0, 0, 0};

    const signed char* aRow = A + (long long)(mbase + l15) * K;
    const signed char* bRow[4];
#pragma unroll
    for (int t = 0; t < 4; ++t)
        bRow[t] = W + (long long)(nbase + t * 16 + l15) * K;

    for (int kb = 0; kb < K; kb += 64) {
        v8i a = load_a_frag(aRow + kb, h);
        __builtin_prefetch(aRow + kb + 512, 0, 1);
        __builtin_prefetch(bRow[0] + kb + 512, 0, 1);
#pragma unroll
        for (int t = 0; t < 4; ++t) {
            v8i b = load_b_frag(bRow[t] + kb, h);
            acc[t] = __builtin_amdgcn_wmma_i32_16x16x64_iu8(
                true, a, true, b, acc[t], false, false);
        }
    }

#pragma unroll
    for (int t = 0; t < 4; ++t) {
        const int n = nbase + t * 16 + l15;
        const float bn = bias[n];
#pragma unroll
        for (int r = 0; r < 8; ++r) {
            const int m = mbase + r + 8 * h;
            float v = (float)acc[t][r] * rs[m] * mW + bn;
            const long long oi = (long long)m * N + n;
            if (MODE == 0) {
                out[oi] = v;
            } else if (MODE == 1) {
                out[oi] = 0.5f * v * (1.0f + erff(v * 0.70710678118654752f));
            } else if (MODE == 2) {
                out[oi] = aux0[oi] + v;
            } else {
                out[oi] = aux0[oi] + aux1[m] * v;
            }
        }
    }
}

extern "C" void kernel_launch(void* const* d_in, const int* in_sizes, int n_in,
                              void* d_out, int out_size, void* d_ws,
                              size_t ws_size, hipStream_t stream) {
    (void)in_sizes; (void)n_in; (void)out_size; (void)ws_size;
    const float* x    = (const float*)d_in[0];
    const float* y    = (const float*)d_in[1];
    const float* z    = (const float*)d_in[2];
    const float* fm   = (const float*)d_in[3];
    const float* w_in = (const float*)d_in[4];
    const float* b_in = (const float*)d_in[5];
    const float* fc1w = (const float*)d_in[6];
    const float* fc1b = (const float*)d_in[7];
    const float* fc2w = (const float*)d_in[8];
    const float* fc2b = (const float*)d_in[9];
    const float* wout = (const float*)d_in[10];
    const float* bout = (const float*)d_in[11];
    float* outp = (float*)d_out;

    char* ws = (char*)d_ws;
    float*       wsum  = (float*)(ws + OFF_WSUM);
    float*       rsbuf = (float*)(ws + OFF_RS);
    signed char* aq    = (signed char*)(ws + OFF_AQ);
    float*       ubuf  = (float*)(ws + OFF_U);
    float*       hbuf  = (float*)(ws + OFF_H);
    signed char* wq_in  = (signed char*)(ws + OFF_WQ);
    signed char* wq_fc1 = wq_in + (long long)D_ * 3 * D_;
    signed char* wq_fc2 = wq_fc1 + 2LL * H_ * D_;
    signed char* wq_out = wq_fc2 + 2LL * D_ * H_;

    const long long NW0 = (long long)D_ * 3 * D_;
    const long long NF  = (long long)H_ * D_;
    const long long NWO = (long long)D_ * D_;

    zero_kernel<<<1, 32, 0, stream>>>(wsum, 8);

    auto red = [&](const float* w, long long n, float* acc) {
        long long b = (n + 255) / 256;
        int blocks = (int)(b < 2048 ? b : 2048);
        absum_kernel<<<blocks, 256, 0, stream>>>(w, n, acc);
    };
    red(w_in,            NW0, wsum + 0);
    red(fc1w + 0 * NF,   NF,  wsum + 1);
    red(fc1w + 1 * NF,   NF,  wsum + 2);
    red(fc2w + 0 * NF,   NF,  wsum + 3);
    red(fc2w + 1 * NF,   NF,  wsum + 4);
    red(wout,            NWO, wsum + 5);

    auto wq = [&](const float* w, signed char* q, long long n, const float* acc) {
        long long b = (n + 255) / 256;
        int blocks = (int)(b < 4096 ? b : 4096);
        wquant_kernel<<<blocks, 256, 0, stream>>>(w, q, n, acc, 1.0f / (float)n);
    };
    wq(w_in,          wq_in,            NW0, wsum + 0);
    wq(fc1w + 0 * NF, wq_fc1 + 0 * NF,  NF,  wsum + 1);
    wq(fc1w + 1 * NF, wq_fc1 + 1 * NF,  NF,  wsum + 2);
    wq(fc2w + 0 * NF, wq_fc2 + 0 * NF,  NF,  wsum + 3);
    wq(fc2w + 1 * NF, wq_fc2 + 1 * NF,  NF,  wsum + 4);
    wq(wout,          wq_out,           NWO, wsum + 5);

    const dim3 blk(256);
    actquant_kernel<3><<<NTOK_, blk, 0, stream>>>(x, y, z, D_, aq, rsbuf);
    gemm_iu8_kernel<0><<<dim3(D_ / 256, NTOK_ / 32), blk, 0, stream>>>(
        aq, wq_in, rsbuf, wsum + 0, 1.0f / (float)NW0, b_in, nullptr, nullptr,
        hbuf, D_, 3 * D_);

    for (int l = 0; l < 2; ++l) {
        actquant_kernel<1><<<NTOK_, blk, 0, stream>>>(hbuf, nullptr, nullptr, D_, aq, rsbuf);
        gemm_iu8_kernel<1><<<dim3(H_ / 256, NTOK_ / 32), blk, 0, stream>>>(
            aq, wq_fc1 + (long long)l * NF, rsbuf, wsum + 1 + l, 1.0f / (float)NF,
            fc1b + (long long)l * H_, nullptr, nullptr, ubuf, H_, D_);
        actquant_kernel<1><<<NTOK_, blk, 0, stream>>>(ubuf, nullptr, nullptr, H_, aq, rsbuf);
        gemm_iu8_kernel<2><<<dim3(D_ / 256, NTOK_ / 32), blk, 0, stream>>>(
            aq, wq_fc2 + (long long)l * NF, rsbuf, wsum + 3 + l, 1.0f / (float)NF,
            fc2b + (long long)l * D_, hbuf, nullptr, hbuf, D_, H_);
    }

    actquant_kernel<1><<<NTOK_, blk, 0, stream>>>(hbuf, nullptr, nullptr, D_, aq, rsbuf);
    gemm_iu8_kernel<3><<<dim3(D_ / 256, NTOK_ / 32), blk, 0, stream>>>(
        aq, wq_out, rsbuf, wsum + 5, 1.0f / (float)NWO, bout, y, fm, outp, D_, D_);
}
